// CustomLayer_13889924235512
// MI455X (gfx1250) — matrix operand order repaired and checked by reading, not compiled, not run on hardware
//
#include <hip/hip_runtime.h>

typedef __attribute__((ext_vector_type(16))) __bf16 v16bf;
typedef __attribute__((ext_vector_type(8)))  __bf16 v8bf;
typedef __attribute__((ext_vector_type(8)))  float  v8f;
typedef __attribute__((ext_vector_type(4)))  int    v4i;

#define LEAK 0.5f
#define GAS __attribute__((address_space(1)))
#define LAS __attribute__((address_space(3)))

constexpr int Bsz = 256;   // batch
constexpr int Tsz = 2048;  // time steps
constexpr int Din = 64;    // input features
constexpr int NR  = 128;   // reservoir size
constexpr int BT  = 16;    // batch tile (WMMA M)
constexpr int WAVES = 8;   // one 16-wide N tile per wave

#if __has_builtin(__builtin_amdgcn_global_load_async_to_lds_b128)
#define HAVE_ASYNC 1
#else
#define HAVE_ASYNC 0
#endif

__device__ __forceinline__ void wait_async0() {
#if __has_builtin(__builtin_amdgcn_s_wait_asynccnt)
  __builtin_amdgcn_s_wait_asynccnt(0);
#else
  asm volatile("s_wait_asynccnt 0" ::: "memory");
#endif
}

__device__ __forceinline__ float fast_tanh(float x) {
#if __has_builtin(__builtin_amdgcn_tanhf)
  return __builtin_amdgcn_tanhf(x);
#else
  float r;
  asm volatile("v_tanh_f32 %0, %1\n\tv_nop\n\tv_nop" : "=v"(r) : "v"(x));
  return r;
#endif
}

__device__ __forceinline__ v8f wmma_bf16(const v16bf& a, const v16bf& b, const v8f& c) {
  return __builtin_amdgcn_wmma_f32_16x16x32_bf16(
      /*neg_a=*/false, a, /*neg_b=*/false, b,
      /*c_mod=*/(short)0, c, /*reuse_a=*/false, /*reuse_b=*/false);
}

__global__ __launch_bounds__(32 * WAVES, 1)
void esn_scan_kernel(const float* __restrict__ x,
                     const float* __restrict__ w_in,
                     const float* __restrict__ b_in,
                     const float* __restrict__ w_res,
                     const float* __restrict__ b_res,
                     float* __restrict__ out) {
  // ping-pong bf16 state tile shared between the 8 waves (8 KB)
  __shared__ __bf16 sstate[2][BT][NR];
#if HAVE_ASYNC
  // ping-pong f32 x_t staging tile filled by async DMA (8 KB)
  __shared__ float xstage[2][BT][Din];
#endif

  const int tid   = threadIdx.x;
  const int nt    = tid >> 5;        // wave id == N-tile index (0..7)
  const int lane  = tid & 31;
  const int m     = lane & 15;       // A-matrix row / C column-within-tile
  const int kh    = lane >> 4;       // lane half (K split for A/B layouts)
  const int n     = nt * 16 + m;     // global reservoir column for B/C
  const int btile = blockIdx.x;      // 16-row batch tile

  const float* xrow = x + (size_t)(btile * BT + m) * Tsz * Din;

#if HAVE_ASYNC
  // one 16-byte async copy slot per thread: 256 threads cover 16x64 f32 tile
  const int arow = tid >> 4;
  const int acol = (tid & 15) * 4;
  const float* agsrc = x + (size_t)(btile * BT + arow) * Tsz * Din + acol;
  auto stage_x = [&](int t, int xb) {
    __builtin_amdgcn_global_load_async_to_lds_b128(
        (GAS v4i*)(agsrc + (size_t)t * Din),
        (LAS v4i*)&xstage[xb][arow][acol], 0, 0);
  };
  stage_x(0, 0);   // overlap t=0 staging with weight-fragment setup
#endif

  // ---------------- persistent weight B-fragments (registers) -------------
  // B 32x16 bf16 layout: element e -> K = kh*8 + e (e<8), 16 + kh*8 + (e-8) (e>=8) ; N = lane&15.
  v16bf bwin[2];
#pragma unroll
  for (int s = 0; s < 2; ++s) {
    const int k0 = s * 32 + kh * 8;
    v16bf f;
#pragma unroll
    for (int e = 0; e < 16; ++e) f[e] = (__bf16)w_in[(size_t)(k0 + e + (e >= 8 ? 8 : 0)) * NR + n];
    bwin[s] = f;
  }
  v16bf bwr[4];
#pragma unroll
  for (int s = 0; s < 4; ++s) {
    const int k0 = s * 32 + kh * 8;
    v16bf f;
#pragma unroll
    for (int e = 0; e < 16; ++e) f[e] = (__bf16)w_res[(size_t)(k0 + e + (e >= 8 ? 8 : 0)) * NR + n];
    bwr[s] = f;
  }
  const float bias0 = b_in[n];            // t = 0 uses b_in only
  const float biasR = bias0 + b_res[n];   // t >= 1 uses b_in + b_res (fused)

  // Build A fragments (16x64 bf16 = two 16x32 frags) from an f32 row.
  // A 16x32 layout: lane half kh gets K chunks [kh*8, kh*8+8) and +16.
  auto load_ax = [&](const float* p, v16bf ax[2]) {
#pragma unroll
    for (int s = 0; s < 2; ++s) {
      const int kb = s * 32 + kh * 8;
      float4 a0 = *(const float4*)(p + kb);
      float4 a1 = *(const float4*)(p + kb + 4);
      float4 c0 = *(const float4*)(p + kb + 16);
      float4 c1 = *(const float4*)(p + kb + 20);
      v16bf f;
      f[0]  = (__bf16)a0.x; f[1]  = (__bf16)a0.y; f[2]  = (__bf16)a0.z; f[3]  = (__bf16)a0.w;
      f[4]  = (__bf16)a1.x; f[5]  = (__bf16)a1.y; f[6]  = (__bf16)a1.z; f[7]  = (__bf16)a1.w;
      f[8]  = (__bf16)c0.x; f[9]  = (__bf16)c0.y; f[10] = (__bf16)c0.z; f[11] = (__bf16)c0.w;
      f[12] = (__bf16)c1.x; f[13] = (__bf16)c1.y; f[14] = (__bf16)c1.z; f[15] = (__bf16)c1.w;
      ax[s] = f;
    }
  };

  // Emit new state: bf16 to LDS first (gates the barrier), then f32 to global.
  // C layout: VGPR r -> M = r + 8*kh ; N = lane&15 (+ tile base).
  auto emit = [&](int t, const v8f& st, int nbuf) {
#pragma unroll
    for (int r = 0; r < 8; ++r) sstate[nbuf][r + 8 * kh][n] = (__bf16)st[r];
#pragma unroll
    for (int r = 0; r < 8; ++r)
      out[((size_t)(btile * BT + r + 8 * kh) * Tsz + t) * NR + n] = st[r];
  };

  const v8f vzero = {};

#if HAVE_ASYNC
  wait_async0();
#endif
  __syncthreads();   // async path: xstage[0] visible to all waves

  // ---------------- t = 0 : x0 = 0.5 * tanh(x@W_in + b_in) ----------------
  v8f st;
  {
#if HAVE_ASYNC
    stage_x(1, 1);   // prefetch t=1 while computing t=0
    v16bf ax[2];
    load_ax(&xstage[0][m][0], ax);
#else
    v16bf ax[2];
    load_ax(xrow, ax);
#endif
    v8f u = wmma_bf16(ax[0], bwin[0], vzero);   // SRC2 = inline 0
    u = wmma_bf16(ax[1], bwin[1], u);
#pragma unroll
    for (int r = 0; r < 8; ++r) st[r] = LEAK * fast_tanh(u[r] + bias0);
    emit(0, st, 0);
  }
#if HAVE_ASYNC
  wait_async0();
#endif
  __syncthreads();

  // ---------------- scan t = 1 .. T-1 -------------------------------------
  for (int t = 1; t < Tsz; ++t) {
    const int sbuf = (t - 1) & 1;

    // independent chain: input projection (no dependence on barrier/LDS state)
    v16bf ax[2];
#if HAVE_ASYNC
    const int xb = t & 1;
    if (t + 1 < Tsz) stage_x(t + 1, xb ^ 1);
    load_ax(&xstage[xb][m][0], ax);
#else
    load_ax(xrow + (size_t)t * Din, ax);
    if (t + 8 < Tsz) __builtin_prefetch(xrow + (size_t)(t + 8) * Din, 0, 0);
#endif
    v8f u = wmma_bf16(ax[0], bwin[0], vzero);
    u = wmma_bf16(ax[1], bwin[1], u);

    // critical chains: previous state (bf16, from LDS) @ W_res, split 2-deep
    v16bf as[4];
#pragma unroll
    for (int s = 0; s < 4; ++s) {
      const __bf16* row = &sstate[sbuf][m][s * 32 + kh * 8];
      v8bf c0 = *(const v8bf*)row;
      v8bf c1 = *(const v8bf*)(row + 16);
      v16bf f;
#pragma unroll
      for (int e = 0; e < 8; ++e) { f[e] = c0[e]; f[8 + e] = c1[e]; }
      as[s] = f;
    }
    v8f r0 = wmma_bf16(as[0], bwr[0], vzero);
    r0 = wmma_bf16(as[1], bwr[1], r0);
    v8f r1 = wmma_bf16(as[2], bwr[2], vzero);
    r1 = wmma_bf16(as[3], bwr[3], r1);

#pragma unroll
    for (int r = 0; r < 8; ++r) {
      const float pre = (u[r] + r0[r]) + (r1[r] + biasR);
      st[r] = (1.0f - LEAK) * st[r] + LEAK * fast_tanh(pre);
    }

    emit(t, st, sbuf ^ 1);
#if HAVE_ASYNC
    wait_async0();
#endif
    __syncthreads();
  }
}

extern "C" void kernel_launch(void* const* d_in, const int* in_sizes, int n_in,
                              void* d_out, int out_size, void* d_ws, size_t ws_size,
                              hipStream_t stream) {
  (void)in_sizes; (void)n_in; (void)out_size; (void)d_ws; (void)ws_size;
  const float* x     = (const float*)d_in[0];
  const float* w_in  = (const float*)d_in[1];
  const float* b_in  = (const float*)d_in[2];
  const float* w_res = (const float*)d_in[3];
  const float* b_res = (const float*)d_in[4];
  float* out = (float*)d_out;

  dim3 grid(Bsz / BT);          // 16 batch tiles
  dim3 block(32 * WAVES);       // 8 waves (wave32)
  hipLaunchKernelGGL(esn_scan_kernel, grid, block, 0, stream,
                     x, w_in, b_in, w_res, b_res, out);
}
